// STTGNN_77129022701970
// MI455X (gfx1250) — matrix operand order repaired and checked by reading, not compiled, not run on hardware
//
#include <hip/hip_runtime.h>

// ---------------------------------------------------------------------------
// STTGNN for MI455X (gfx1250, wave32, WMMA).
//  - All dense GEMMs on v_wmma_f32_16x16x32_bf16 (f32 accumulate).
//  - GRU gate math runs directly on WMMA accumulator registers (r/z/xn/hn
//    tiles share the (lane,vgpr)<->(m,k) mapping); previous h is read from
//    LDS at the matching C-layout position, so no register array is needed
//    (keeps allocation < 256 VGPRs -> no s_set_vgpr_msb churn).
//  - Activations use native v_exp_f32 + v_rcp_f32 (no IEEE divide expansion).
//  - Packed bf16 weight fragments (48 KB, contiguous) are DMA'd into LDS once
//    per block with the Tensor Data Mover; B operands are ds_load_b128 pairs.
//  - Edge gather/scatter (the 23.3 TB/s-bound part) uses coalesced float4
//    gathers + global f32 atomic adds, with global_prefetch warm-ups.
// ---------------------------------------------------------------------------

typedef __attribute__((ext_vector_type(16))) __bf16        v16bf;
typedef __attribute__((ext_vector_type(8)))  float         v8f;
typedef __attribute__((ext_vector_type(4)))  unsigned int  u32x4;
typedef __attribute__((ext_vector_type(8)))  int           i32x8;
typedef __attribute__((ext_vector_type(4)))  int           i32x4;

#define T_   24
#define DYN_ 16
#define H_   64

__device__ __forceinline__ v8f wmma_bf16(v16bf a, v16bf b, v8f c) {
  return __builtin_amdgcn_wmma_f32_16x16x32_bf16(false, a, false, b, (short)0, c,
                                                 false, false);
}

// sigmoid(x) = rcp(1 + exp(-x))           (v_exp_f32 + v_rcp_f32)
__device__ __forceinline__ float fast_sigmoid(float x) {
  return __builtin_amdgcn_rcpf(1.0f + __expf(-x));
}
// tanh(x) = 2*sigmoid(2x) - 1; saturates correctly at +-1 for large |x|
__device__ __forceinline__ float fast_tanh(float x) {
  float s = __builtin_amdgcn_rcpf(1.0f + __expf(-2.0f * x));
  return 2.0f * s - 1.0f;
}

// Issue a 1-D TDM copy of `dwords` dwords from global `gaddr` to LDS `laddr`.
__device__ __forceinline__ void tdm_copy_1d(unsigned long long gaddr,
                                            unsigned int laddr, int dwords) {
  u32x4 g0;
  g0[0] = 1u;                                              // count = 1
  g0[1] = laddr;                                           // lds_addr
  g0[2] = (unsigned int)(gaddr & 0xffffffffu);             // global_addr lo
  g0[3] = (unsigned int)((gaddr >> 32) & 0x1ffffffu) | (2u << 30);  // hi | type=2
  i32x8 g1;
  g1[0] = (2 << 16);                            // data_size = 4 bytes
  g1[1] = (int)((dwords & 0xffff) << 16);       // tensor_dim0 lo16
  g1[2] = (int)((dwords >> 16) | (1 << 16));    // tensor_dim0 hi | tensor_dim1=1
  g1[3] = (int)((dwords & 0xffff) << 16);       // tile_dim0
  g1[4] = 1;                                    // tile_dim1 = 1
  g1[5] = dwords;                               // tensor_dim0_stride
  g1[6] = 0;
  g1[7] = 0;
  i32x4 z4 = {0, 0, 0, 0};
#if __has_include(<hip/amd_detail/amd_gfx1250_TDM.h>)
  i32x8 z8 = {0, 0, 0, 0, 0, 0, 0, 0};
  __builtin_amdgcn_tensor_load_to_lds(g0, g1, z4, z4, z8, 0);
#else
  __builtin_amdgcn_tensor_load_to_lds(g0, g1, z4, z4, 0);
#endif
}

// ---------------------------------------------------------------------------
// Pre-pack all dense weights into bf16 WMMA B-fragments (ISA B layout:
// element (lane,j) of a 32-K chunk = W[kc*32 + 16*(j>>3) + 8*(lane>>4) + (j&7)][n0+(lane&15)]).
// Order: Wih 12 frags, Whh 24, W1 12, W2 8  (56 * 512 bf16 = 56 KB).
// ---------------------------------------------------------------------------
__global__ void sttgnn_prepack_kernel(const float* __restrict__ Wih,
                                      const float* __restrict__ Whh,
                                      const float* __restrict__ W1,
                                      const float* __restrict__ W2,
                                      __bf16* __restrict__ frags) {
  int tid = blockIdx.x * blockDim.x + threadIdx.x;
  if (tid >= 56 * 512) return;
  int frag = tid >> 9;
  int e    = tid & 511;
  int lane = e >> 4;
  int j    = e & 15;
  int n    = lane & 15;
  int k    = (j & 7) + 8 * (lane >> 4) + 16 * (j >> 3);   // 0..31 inside K-chunk (B layout)
  float v = 0.0f;
  if (frag < 12) {                    // Wih [192,16]: B[k][g] = Wih[g][k]
    int g = frag * 16 + n;
    if (k < DYN_) v = Wih[g * DYN_ + k];
  } else if (frag < 36) {             // Whh [192,64]: B[k][g] = Whh[g][k]
    int f = frag - 12, nt = f >> 1, kc = f & 1;
    v = Whh[(nt * 16 + n) * H_ + kc * 32 + k];
  } else if (frag < 48) {             // W1 [72,64]: B[k][c] = W1[k][c]
    int f = frag - 36, nt = f / 3, kc = f % 3;
    int kk = kc * 32 + k;
    if (kk < H_ + 8) v = W1[kk * H_ + nt * 16 + n];
  } else {                            // W2 [64,64]
    int f = frag - 48, nt = f >> 1, kc = f & 1;
    v = W2[(kc * 32 + k) * H_ + nt * 16 + n];
  }
  frags[tid] = (__bf16)v;
}

// ---------------------------------------------------------------------------
// Fused GRU (24 steps) + layer-1 GEMM. 2 waves per block, one 16-node tile
// per wave; weight fragments TDM'd into LDS once per block.
// ---------------------------------------------------------------------------
__global__ __launch_bounds__(64) void sttgnn_gru_w1_kernel(
    const float* __restrict__ x_dyn, const float* __restrict__ x_stat,
    const float* __restrict__ bih, const float* __restrict__ bhh,
    const __bf16* __restrict__ frags, float* __restrict__ hw1, int ntiles) {
  __shared__ __align__(16) __bf16 wfr[48 * 512];   // 48 KB packed weights
  __shared__ float hbuf[2][16 * H_];               // per-wave h state
  __shared__ float sbrz[128], sbnx[H_], sbnh[H_];

  const int wid  = (int)threadIdx.x >> 5;
  const int lane = (int)threadIdx.x & 31;
  const int l15  = lane & 15;
  const int half = lane >> 4;

  if (threadIdx.x < 32) {  // wave 0: DMA the weight fragments (12288 dwords)
    tdm_copy_1d((unsigned long long)(const void*)frags,
                (unsigned int)(unsigned long long)(const void*)&wfr[0],
                48 * 512 / 2);
  }
  for (int i = threadIdx.x; i < 128; i += 64) sbrz[i] = bih[i] + bhh[i];
  for (int i = threadIdx.x; i < H_; i += 64) {
    sbnx[i] = bih[128 + i];
    sbnh[i] = bhh[128 + i];
  }
  for (int i = threadIdx.x; i < 2 * 16 * H_; i += 64) (&hbuf[0][0])[i] = 0.0f;
  if (threadIdx.x < 32) __builtin_amdgcn_s_wait_tensorcnt(0);
  __syncthreads();

  const int tile = blockIdx.x * 2 + wid;
  if (tile >= ntiles) return;              // wave-uniform; EXEC stays full below
  const int n0 = tile * 16;
  float* hb = hbuf[wid];
  const v16bf* wv = (const v16bf*)wfr;
  const v8f vzero = {0.f, 0.f, 0.f, 0.f, 0.f, 0.f, 0.f, 0.f};

  // this lane's contiguous x slice: row n0+l15, columns 8*half..8*half+7
  const float* xrow = x_dyn + (size_t)(n0 + l15) * (T_ * DYN_) + 8 * half;

#pragma unroll 1
  for (int t = 0; t < T_; ++t) {
    // A fragments (ISA 16-bit A 16x32 layout): per lane two contiguous 8-float
    // runs at k = 8*half and k = 16 + 8*half of each 32-K chunk.
    v16bf ax, ah0, ah1;
#pragma unroll
    for (int j = 0; j < 8; ++j) {
      ax[j]      = (__bf16)xrow[t * DYN_ + j];
      ax[8 + j]  = (__bf16)0.0f;
      ah0[j]     = (__bf16)hb[l15 * H_ + 8 * half + j];
      ah0[8 + j] = (__bf16)hb[l15 * H_ + 16 + 8 * half + j];
      ah1[j]     = (__bf16)hb[l15 * H_ + 32 + 8 * half + j];
      ah1[8 + j] = (__bf16)hb[l15 * H_ + 48 + 8 * half + j];
    }
    // gates per hidden k-tile; r/z/xn/hn accumulators share (lane,i) <-> (m,k),
    // and hb[(i+8*half)*64 + kt*16 + l15] is h_old at the same position.
#pragma unroll 1
    for (int kt = 0; kt < 4; ++kt) {
      v8f accr = wmma_bf16(ax, wv[kt * 32 + lane], vzero);
      accr = wmma_bf16(ah0, wv[(12 + 2 * kt + 0) * 32 + lane], accr);
      accr = wmma_bf16(ah1, wv[(12 + 2 * kt + 1) * 32 + lane], accr);
      v8f accz = wmma_bf16(ax, wv[(4 + kt) * 32 + lane], vzero);
      accz = wmma_bf16(ah0, wv[(20 + 2 * kt + 0) * 32 + lane], accz);
      accz = wmma_bf16(ah1, wv[(20 + 2 * kt + 1) * 32 + lane], accz);
      v8f accxn = wmma_bf16(ax, wv[(8 + kt) * 32 + lane], vzero);
      v8f acchn = wmma_bf16(ah0, wv[(28 + 2 * kt + 0) * 32 + lane], vzero);
      acchn = wmma_bf16(ah1, wv[(28 + 2 * kt + 1) * 32 + lane], acchn);

      const int k = kt * 16 + l15;
      const float br = sbrz[k], bz = sbrz[64 + k];
      const float bx = sbnx[k], bh = sbnh[k];
#pragma unroll
      for (int i = 0; i < 8; ++i) {
        const int hidx = (i + 8 * half) * H_ + k;
        float r  = fast_sigmoid(accr[i] + br);
        float zg = fast_sigmoid(accz[i] + bz);
        float ng = fast_tanh(accxn[i] + bx + r * (acchn[i] + bh));
        float ho = hb[hidx];
        hb[hidx] = fmaf(zg, ho - ng, ng);   // (1-z)*n + z*h
      }
    }
  }

  // ---- fused layer-1 GEMM: hw1 = [h | x_stat] @ W1, K = 72 padded to 3x32 ----
  v16bf a0, a1, a2s;
#pragma unroll
  for (int j = 0; j < 8; ++j) {
    a0[j]      = (__bf16)hb[l15 * H_ + 8 * half + j];
    a0[8 + j]  = (__bf16)hb[l15 * H_ + 16 + 8 * half + j];
    a1[j]      = (__bf16)hb[l15 * H_ + 32 + 8 * half + j];
    a1[8 + j]  = (__bf16)hb[l15 * H_ + 48 + 8 * half + j];
    // chunk 2: k = 64 + kl; only kl < 8 (x_stat) nonzero -> j<8, half==0 lanes
    a2s[j]     = (__bf16)((half == 0) ? x_stat[(size_t)(n0 + l15) * 8 + j] : 0.0f);
    a2s[8 + j] = (__bf16)0.0f;
  }
#pragma unroll 1
  for (int nt = 0; nt < 4; ++nt) {
    v8f acc = wmma_bf16(a0, wv[(36 + 3 * nt + 0) * 32 + lane], vzero);
    acc = wmma_bf16(a1, wv[(36 + 3 * nt + 1) * 32 + lane], acc);
    acc = wmma_bf16(a2s, wv[(36 + 3 * nt + 2) * 32 + lane], acc);
#pragma unroll
    for (int i = 0; i < 8; ++i)
      hw1[(size_t)(n0 + i + 8 * half) * H_ + nt * 16 + l15] = acc[i];
  }
}

// ---------------------------------------------------------------------------
// Layer-2 GEMM: hw2 = relu(agg1) @ W2 (K=64, 2 chunks), 4 waves per block.
// ---------------------------------------------------------------------------
__global__ __launch_bounds__(128) void sttgnn_gcn2_kernel(
    const float* __restrict__ agg1, const __bf16* __restrict__ fragW2,
    float* __restrict__ hw2, int ntiles) {
  const int wid  = (int)threadIdx.x >> 5;
  const int lane = (int)threadIdx.x & 31;
  const int l15  = lane & 15;
  const int half = lane >> 4;
  const int tile = blockIdx.x * 4 + wid;
  if (tile >= ntiles) return;              // wave-uniform
  const int n0 = tile * 16;
  const v16bf* wv = (const v16bf*)fragW2;
  const v8f vzero = {0.f, 0.f, 0.f, 0.f, 0.f, 0.f, 0.f, 0.f};
  const float* row = agg1 + (size_t)(n0 + l15) * H_ + 8 * half;
  v16bf a0, a1;
#pragma unroll
  for (int j = 0; j < 8; ++j) {
    a0[j]     = (__bf16)fmaxf(row[j], 0.0f);
    a0[8 + j] = (__bf16)fmaxf(row[16 + j], 0.0f);
    a1[j]     = (__bf16)fmaxf(row[32 + j], 0.0f);
    a1[8 + j] = (__bf16)fmaxf(row[48 + j], 0.0f);
  }
#pragma unroll 1
  for (int nt = 0; nt < 4; ++nt) {
    v8f acc = wmma_bf16(a0, wv[(2 * nt + 0) * 32 + lane], vzero);
    acc = wmma_bf16(a1, wv[(2 * nt + 1) * 32 + lane], acc);
#pragma unroll
    for (int i = 0; i < 8; ++i)
      hw2[(size_t)(n0 + i + 8 * half) * H_ + nt * 16 + l15] = acc[i];
  }
}

// agg[n][k] = b[k]
__global__ void sttgnn_init_bias_kernel(float* __restrict__ agg,
                                        const float* __restrict__ b, int total) {
  int i = blockIdx.x * blockDim.x + threadIdx.x;
  if (i < total) agg[i] = b[i & (H_ - 1)];
}

// scatter-add: agg[dst] += h[src] * w   (16 float4 chunks per edge)
__global__ void sttgnn_scatter_kernel(const float* __restrict__ hsrc,
                                      const int* __restrict__ src,
                                      const int* __restrict__ dst,
                                      const float* __restrict__ w,
                                      float* __restrict__ agg, int nedge) {
  int total = nedge * 16;
  int step = gridDim.x * blockDim.x;
  for (int i = blockIdx.x * blockDim.x + threadIdx.x; i < total; i += step) {
    int e = i >> 4;
    int c = i & 15;
    int s = src[e], d = dst[e];
    float ww = w[e];
    if (c == 0) {  // warm a future gather row into cache
      int e2 = e + 32;
      if (e2 < nedge) __builtin_prefetch(hsrc + (size_t)src[e2] * H_, 0, 1);
    }
    float4 v = ((const float4*)(hsrc + (size_t)s * H_))[c];
    float* ap = agg + (size_t)d * H_ + c * 4;
    atomicAdd(ap + 0, v.x * ww);
    atomicAdd(ap + 1, v.y * ww);
    atomicAdd(ap + 2, v.z * ww);
    atomicAdd(ap + 3, v.w * ww);
  }
}

// heads: aqi = relu(h)@Wa + ba ; pm = relu(h)@Wp + bp
__global__ void sttgnn_heads_kernel(const float* __restrict__ agg2,
                                    const float* __restrict__ Wa,
                                    const float* __restrict__ ba,
                                    const float* __restrict__ Wp,
                                    const float* __restrict__ bp,
                                    float* __restrict__ out, int n) {
  int i = blockIdx.x * blockDim.x + threadIdx.x;
  if (i >= n) return;
  float sa = ba[0], sp = bp[0];
#pragma unroll 8
  for (int k = 0; k < H_; ++k) {
    float h = fmaxf(agg2[(size_t)i * H_ + k], 0.0f);
    sa += h * Wa[k];
    sp += h * Wp[k];
  }
  out[i] = sa;
  out[(size_t)n + i] = sp;
}

extern "C" void kernel_launch(void* const* d_in, const int* in_sizes, int n_in,
                              void* d_out, int out_size, void* d_ws, size_t ws_size,
                              hipStream_t stream) {
  const float* x_dyn  = (const float*)d_in[0];
  const float* x_stat = (const float*)d_in[1];
  const int*   eidx   = (const int*)d_in[2];
  const float* ew     = (const float*)d_in[3];
  const float* Wih    = (const float*)d_in[4];
  const float* Whh    = (const float*)d_in[5];
  const float* bih    = (const float*)d_in[6];
  const float* bhh    = (const float*)d_in[7];
  const float* W1     = (const float*)d_in[8];
  const float* b1     = (const float*)d_in[9];
  const float* W2     = (const float*)d_in[10];
  const float* b2     = (const float*)d_in[11];
  const float* Wa     = (const float*)d_in[12];
  const float* ba     = (const float*)d_in[13];
  const float* Wp     = (const float*)d_in[14];
  const float* bp     = (const float*)d_in[15];

  const int N_ = in_sizes[1] / 8;    // 50000
  const int E_ = in_sizes[2] / 2;    // 1.6M
  const int ntiles = (N_ + 15) / 16; // 3125
  const int* src = eidx;
  const int* dst = eidx + E_;

  char* ws = (char*)d_ws;
  __bf16* frags = (__bf16*)ws;                                   // 56 KB packed weights
  float*  bufA  = (float*)(ws + 65536);                          // hw1 / hw2
  float*  bufB  = (float*)(ws + 65536 + (size_t)N_ * H_ * 4);    // agg1 / agg2

  sttgnn_prepack_kernel<<<(56 * 512 + 255) / 256, 256, 0, stream>>>(Wih, Whh, W1, W2,
                                                                    frags);
  sttgnn_gru_w1_kernel<<<(ntiles + 1) / 2, 64, 0, stream>>>(x_dyn, x_stat, bih, bhh,
                                                            frags, bufA, ntiles);
  sttgnn_init_bias_kernel<<<(N_ * H_ + 255) / 256, 256, 0, stream>>>(bufB, b1, N_ * H_);
  sttgnn_scatter_kernel<<<2048, 256, 0, stream>>>(bufA, src, dst, ew, bufB, E_);
  sttgnn_gcn2_kernel<<<(ntiles + 3) / 4, 128, 0, stream>>>(bufB, frags + 48 * 512, bufA,
                                                           ntiles);
  sttgnn_init_bias_kernel<<<(N_ * H_ + 255) / 256, 256, 0, stream>>>(bufB, b2, N_ * H_);
  sttgnn_scatter_kernel<<<2048, 256, 0, stream>>>(bufA, src, dst, ew, bufB, E_);
  sttgnn_heads_kernel<<<(N_ + 255) / 256, 256, 0, stream>>>(bufB, Wa, ba, Wp, bp,
                                                            (float*)d_out, N_);
}
